// Decoder_33870112096565
// MI455X (gfx1250) — matrix operand order repaired and checked by reading, not compiled, not run on hardware
//
#include <hip/hip_runtime.h>

// ---------------------------------------------------------------------------
// GRU decoder step, CDNA5 (gfx1250) wave32 WMMA bf16 implementation.
// B=32768, H=1024, E=512.  Compute-bound (~722 GFLOP): bf16 WMMA, f32 accum.
// Round 2: weights packed K-major (kills ds_store_b16 transpose scatter),
// double-buffered LDS (one barrier / K-step, global loads overlap WMMA).
// ---------------------------------------------------------------------------

typedef __attribute__((ext_vector_type(16))) __bf16 v16bf;
typedef __attribute__((ext_vector_type(8)))  float  v8f;

#define BDIM 32768
#define HDIM 1024
#define EDIM 512
#define KTOT 3584   // E + H + 2H
#define NTOT 3072   // [z | r | w] column groups
#define LDT  40     // LDS row pitch in halves (32 + 8 pad, 80B = 16B-aligned)

// f32 -> bf16 round-to-nearest-even (raw u16)
__device__ __forceinline__ unsigned short f2bu(float f) {
  unsigned u = __builtin_bit_cast(unsigned, f);
  u += 0x7FFFu + ((u >> 16) & 1u);
  return (unsigned short)(u >> 16);
}
__device__ __forceinline__ float bu2f(unsigned short h) {
  unsigned u = ((unsigned)h) << 16;
  return __builtin_bit_cast(float, u);
}
__device__ __forceinline__ float sigmoidf_fast(float x) {
  return 1.0f / (1.0f + __expf(-x));
}
__device__ __forceinline__ float tanhf_fast(float x) {
  float e2 = __expf(2.0f * x);
  return 1.0f - 2.0f / (e2 + 1.0f);
}

struct B32B { uint4 lo, hi; };
// Build a 16-element bf16 fragment from two 16B-aligned chunks of 8 halves.
__device__ __forceinline__ v16bf frag_from(const unsigned short* p0,
                                           const unsigned short* p1) {
  B32B t;
  t.lo = *(const uint4*)p0;
  t.hi = *(const uint4*)p1;
  return __builtin_bit_cast(v16bf, t);
}

// ---------------------------------------------------------------------------
// Weight packing, K-major ("transposed") so GEMM B-tiles stage with b128 ops:
//   WbigT[c][k], c in [0,3072), k in [0,3584):
//     k<512        : (Wz|Wr|W)[k][c%1024]  by column group c/1024
//     512<=k<1536  : (Uz|Ur|0)[k-512][c%1024]   (U handled in pass 2)
//     else         : (Cz|Cr|C)[k-1536][c%1024]
//   UbT[c][k] = U[k][c]
// One-time pass over 44 MB; output lives in the 192 MB L2 afterwards.
// ---------------------------------------------------------------------------
__global__ void __launch_bounds__(256)
pack_w1t_kernel(const float* __restrict__ Wz, const float* __restrict__ Wr,
                const float* __restrict__ W,  const float* __restrict__ Uz,
                const float* __restrict__ Ur, const float* __restrict__ Cz,
                const float* __restrict__ Cr, const float* __restrict__ C,
                unsigned short* __restrict__ WbigT) {
  int idx = blockIdx.x * 256 + threadIdx.x;   // output-linear: c*KTOT + k
  if (idx >= KTOT * NTOT) return;
  int c = idx / KTOT;
  int k = idx - c * KTOT;
  int g = c >> 10;
  int cc = c & 1023;
  float v;
  if (k < EDIM) {
    const float* p = (g == 0) ? Wz : ((g == 1) ? Wr : W);
    v = p[k * HDIM + cc];
  } else if (k < EDIM + HDIM) {
    int kk = k - EDIM;
    v = (g == 0) ? Uz[kk * HDIM + cc]
                 : ((g == 1) ? Ur[kk * HDIM + cc] : 0.0f);
  } else {
    int kk = k - EDIM - HDIM;
    const float* p = (g == 0) ? Cz : ((g == 1) ? Cr : C);
    v = p[kk * HDIM + cc];
  }
  WbigT[idx] = f2bu(v);
}

__global__ void __launch_bounds__(256)
pack_ut_kernel(const float* __restrict__ U, unsigned short* __restrict__ UbT) {
  int idx = blockIdx.x * 256 + threadIdx.x;   // c*HDIM + k
  if (idx >= HDIM * HDIM) return;
  int c = idx >> 10;
  int k = idx & 1023;
  UbT[idx] = f2bu(U[k * HDIM + c]);
}

// ---------------------------------------------------------------------------
// GEMM1: P = [emb[y] | s | c] (B x 3584, bf16 on the fly) x Wbig (3584 x 3072)
// Fused epilogue per 1024-wide column group:
//   g0: zbuf  = bf16(sigmoid(P))
//   g1: rsbuf = bf16(sigmoid(P) * s)
//   g2: awbuf = P (f32)
// 128x128 tile / 4-wave block; each wave: 64x64 quadrant = 16 WMMA / K-step.
// Double-buffered LDS: stage tile kt+1 while computing tile kt.
// ---------------------------------------------------------------------------
__global__ void __launch_bounds__(128)
gemm1_kernel(const float* __restrict__ s_in, const int* __restrict__ yidx,
             const float* __restrict__ c_in, const float* __restrict__ emb,
             const unsigned short* __restrict__ WbigT,
             unsigned short* __restrict__ zbuf,
             unsigned short* __restrict__ rsbuf,
             float* __restrict__ awbuf) {
  __shared__ unsigned short ldsA[2][128 * LDT];   // [buf][row][k]
  __shared__ unsigned short ldsB[2][128 * LDT];   // [buf][col][k] (K-major)

  const int tid  = threadIdx.x;
  const int lane = tid & 31;
  const int wave = tid >> 5;
  const int rowBase = blockIdx.x * 128;
  const int colBase = blockIdx.y * 128;

  // A loader: thread t owns LDS row t (32 k-values / step), gather + convert
  const int growA = rowBase + tid;
  const int yi = yidx[growA];
  const float* embRow = emb + (long long)yi * EDIM;
  const float* sRow   = s_in + (long long)growA * HDIM;
  const float* cRow   = c_in + (long long)growA * (2 * HDIM);
  // B loader: thread t owns LDS col row t; WbigT row = contiguous k
  const unsigned short* bRow = WbigT + (long long)(colBase + tid) * KTOT;

  const int mQ = (wave >> 1) * 64;
  const int nQ = (wave & 1) * 64;

  v8f acc[4][4];
#pragma unroll
  for (int i = 0; i < 4; ++i)
#pragma unroll
    for (int j = 0; j < 4; ++j) acc[i][j] = (v8f)(0.0f);

  auto stageA = [&](int kt, int buf) {
    const int k0 = kt * 32;
    const float* src;
    if (k0 < EDIM)             src = embRow + k0;
    else if (k0 < EDIM + HDIM) src = sRow + (k0 - EDIM);
    else                       src = cRow + (k0 - EDIM - HDIM);
    __align__(16) unsigned short tmp[32];
#pragma unroll
    for (int j = 0; j < 32; j += 4) {
      float4 f = *(const float4*)(src + j);
      tmp[j + 0] = f2bu(f.x); tmp[j + 1] = f2bu(f.y);
      tmp[j + 2] = f2bu(f.z); tmp[j + 3] = f2bu(f.w);
    }
    uint4* dst = (uint4*)&ldsA[buf][tid * LDT];
    const uint4* sv = (const uint4*)tmp;
    dst[0] = sv[0]; dst[1] = sv[1]; dst[2] = sv[2]; dst[3] = sv[3];
  };
  auto stageB = [&](int kt, int buf) {
    const uint4* sv = (const uint4*)(bRow + kt * 32);
    if (kt + 2 < KTOT / 32)
      __builtin_prefetch(bRow + (kt + 2) * 32, 0, 1);  // global_prefetch_b8
    uint4* dst = (uint4*)&ldsB[buf][tid * LDT];
    dst[0] = sv[0]; dst[1] = sv[1]; dst[2] = sv[2]; dst[3] = sv[3];
  };

  stageA(0, 0);
  stageB(0, 0);

  const int aoff = (lane < 16) ? 0 : 8;    // A frag: halves {k..k+7, k+16..k+23}
  const int boff = (lane < 16) ? 0 : 8;    // B frag: halves {k..k+7, k+16..k+23}

  for (int kt = 0; kt < KTOT / 32; ++kt) {
    __syncthreads();                       // buf cur fully staged
    const int cur = kt & 1;
    if (kt + 1 < KTOT / 32) {              // overlap next-tile staging
      stageA(kt + 1, cur ^ 1);
      stageB(kt + 1, cur ^ 1);
    }
    v16bf afr[4], bfr[4];
#pragma unroll
    for (int mt = 0; mt < 4; ++mt) {
      const unsigned short* ap =
          &ldsA[cur][(mQ + mt * 16 + (lane & 15)) * LDT + aoff];
      afr[mt] = frag_from(ap, ap + 16);
    }
#pragma unroll
    for (int nt = 0; nt < 4; ++nt) {
      const unsigned short* bp =
          &ldsB[cur][(nQ + nt * 16 + (lane & 15)) * LDT + boff];
      bfr[nt] = frag_from(bp, bp + 16);
    }
#pragma unroll
    for (int mt = 0; mt < 4; ++mt)
#pragma unroll
      for (int nt = 0; nt < 4; ++nt)
        acc[mt][nt] = __builtin_amdgcn_wmma_f32_16x16x32_bf16(
            false, afr[mt], false, bfr[nt], (short)0, acc[mt][nt], false, false);
  }

  // ---- fused epilogue (column group is uniform per block: 1024 % 128 == 0)
  const int group = colBase >> 10;
#pragma unroll
  for (int mt = 0; mt < 4; ++mt) {
#pragma unroll
    for (int nt = 0; nt < 4; ++nt) {
      const int gcol  = colBase + nQ + nt * 16 + (lane & 15);
      const int rbase = rowBase + mQ + mt * 16 + ((lane < 16) ? 0 : 8);
#pragma unroll
      for (int v = 0; v < 8; ++v) {
        const long long grow = rbase + v;
        const float val = acc[mt][nt][v];
        if (group == 0) {
          zbuf[grow * HDIM + gcol] = f2bu(sigmoidf_fast(val));
        } else if (group == 1) {
          const int hc = gcol - HDIM;
          const float r = sigmoidf_fast(val);
          rsbuf[grow * HDIM + hc] = f2bu(r * s_in[grow * HDIM + hc]);
        } else {
          awbuf[grow * HDIM + (gcol - 2 * HDIM)] = val;
        }
      }
    }
  }
}

// ---------------------------------------------------------------------------
// GEMM2: s~pre = (r*s) @ U + aw ; s_t = (1-z)*s + z*tanh(s~pre)  -> d_out
// ---------------------------------------------------------------------------
__global__ void __launch_bounds__(128)
gemm2_kernel(const float* __restrict__ s_in,
             const unsigned short* __restrict__ rsbuf,
             const unsigned short* __restrict__ UbT,
             const unsigned short* __restrict__ zbuf,
             const float* __restrict__ awbuf,
             float* __restrict__ out) {
  __shared__ unsigned short ldsA[2][128 * LDT];
  __shared__ unsigned short ldsB[2][128 * LDT];

  const int tid  = threadIdx.x;
  const int lane = tid & 31;
  const int wave = tid >> 5;
  const int rowBase = blockIdx.x * 128;
  const int colBase = blockIdx.y * 128;

  const unsigned short* aRow = rsbuf + (long long)(rowBase + tid) * HDIM;
  const unsigned short* bRow = UbT + (long long)(colBase + tid) * HDIM;

  const int mQ = (wave >> 1) * 64;
  const int nQ = (wave & 1) * 64;

  v8f acc[4][4];
#pragma unroll
  for (int i = 0; i < 4; ++i)
#pragma unroll
    for (int j = 0; j < 4; ++j) acc[i][j] = (v8f)(0.0f);

  auto stage = [&](int kt, int buf) {
    const uint4* sa = (const uint4*)(aRow + kt * 32);
    const uint4* sb = (const uint4*)(bRow + kt * 32);
    if (kt + 2 < HDIM / 32)
      __builtin_prefetch(bRow + (kt + 2) * 32, 0, 1);
    uint4* da = (uint4*)&ldsA[buf][tid * LDT];
    uint4* db = (uint4*)&ldsB[buf][tid * LDT];
    da[0] = sa[0]; da[1] = sa[1]; da[2] = sa[2]; da[3] = sa[3];
    db[0] = sb[0]; db[1] = sb[1]; db[2] = sb[2]; db[3] = sb[3];
  };

  stage(0, 0);

  const int aoff = (lane < 16) ? 0 : 8;
  const int boff = (lane < 16) ? 0 : 8;

  for (int kt = 0; kt < HDIM / 32; ++kt) {
    __syncthreads();
    const int cur = kt & 1;
    if (kt + 1 < HDIM / 32) stage(kt + 1, cur ^ 1);

    v16bf afr[4], bfr[4];
#pragma unroll
    for (int mt = 0; mt < 4; ++mt) {
      const unsigned short* ap =
          &ldsA[cur][(mQ + mt * 16 + (lane & 15)) * LDT + aoff];
      afr[mt] = frag_from(ap, ap + 16);
    }
#pragma unroll
    for (int nt = 0; nt < 4; ++nt) {
      const unsigned short* bp =
          &ldsB[cur][(nQ + nt * 16 + (lane & 15)) * LDT + boff];
      bfr[nt] = frag_from(bp, bp + 16);
    }
#pragma unroll
    for (int mt = 0; mt < 4; ++mt)
#pragma unroll
      for (int nt = 0; nt < 4; ++nt)
        acc[mt][nt] = __builtin_amdgcn_wmma_f32_16x16x32_bf16(
            false, afr[mt], false, bfr[nt], (short)0, acc[mt][nt], false, false);
  }

#pragma unroll
  for (int mt = 0; mt < 4; ++mt) {
#pragma unroll
    for (int nt = 0; nt < 4; ++nt) {
      const int gcol  = colBase + nQ + nt * 16 + (lane & 15);
      const int rbase = rowBase + mQ + mt * 16 + ((lane < 16) ? 0 : 8);
#pragma unroll
      for (int v = 0; v < 8; ++v) {
        const long long idx = (long long)(rbase + v) * HDIM + gcol;
        const float val = acc[mt][nt][v] + awbuf[idx];
        const float st = tanhf_fast(val);
        const float z  = bu2f(zbuf[idx]);
        const float so = s_in[idx];
        out[idx] = (1.0f - z) * so + z * st;
      }
    }
  }
}

// ---------------------------------------------------------------------------
// Host-side launch.  Workspace layout (bytes):
//   awbuf f32  : 134,217,728
//   zbuf  bf16 :  67,108,864
//   rsbuf bf16 :  67,108,864
//   WbigT bf16 :  22,020,096
//   UbT   bf16 :   2,097,152   (total ~292.6 MB)
// ---------------------------------------------------------------------------
extern "C" void kernel_launch(void* const* d_in, const int* in_sizes, int n_in,
                              void* d_out, int out_size, void* d_ws, size_t ws_size,
                              hipStream_t stream) {
  const float* s_in = (const float*)d_in[0];
  const int*   yid  = (const int*)d_in[1];
  const float* c_in = (const float*)d_in[2];
  const float* emb  = (const float*)d_in[3];
  const float* Wz   = (const float*)d_in[4];
  const float* Uz   = (const float*)d_in[5];
  const float* Cz   = (const float*)d_in[6];
  const float* Wr   = (const float*)d_in[7];
  const float* Ur   = (const float*)d_in[8];
  const float* Cr   = (const float*)d_in[9];
  const float* W    = (const float*)d_in[10];
  const float* U    = (const float*)d_in[11];
  const float* C    = (const float*)d_in[12];
  float* out = (float*)d_out;

  char* ws = (char*)d_ws;
  size_t off = 0;
  float* awbuf = (float*)(ws + off);                   off += (size_t)BDIM * HDIM * 4;
  unsigned short* zbuf  = (unsigned short*)(ws + off); off += (size_t)BDIM * HDIM * 2;
  unsigned short* rsbuf = (unsigned short*)(ws + off); off += (size_t)BDIM * HDIM * 2;
  unsigned short* WbigT = (unsigned short*)(ws + off); off += (size_t)KTOT * NTOT * 2;
  unsigned short* UbT   = (unsigned short*)(ws + off);

  // 1) pack weights to bf16, K-major (L2-resident afterwards)
  {
    int n = KTOT * NTOT;
    pack_w1t_kernel<<<(n + 255) / 256, 256, 0, stream>>>(Wz, Wr, W, Uz, Ur, Cz,
                                                         Cr, C, WbigT);
    int m = HDIM * HDIM;
    pack_ut_kernel<<<(m + 255) / 256, 256, 0, stream>>>(U, UbT);
  }
  // 2) big fused GEMM: gates + candidate partial
  {
    dim3 grid(BDIM / 128, NTOT / 128);
    gemm1_kernel<<<grid, 128, 0, stream>>>(s_in, yid, c_in, emb, WbigT, zbuf,
                                           rsbuf, awbuf);
  }
  // 3) (r*s)@U + aw, tanh, final blend
  {
    dim3 grid(BDIM / 128, HDIM / 128);
    gemm2_kernel<<<grid, 128, 0, stream>>>(s_in, rsbuf, UbT, zbuf, awbuf, out);
  }
}
